// Rel_ds_46574625358081
// MI455X (gfx1250) — matrix operand order repaired and checked by reading, not compiled, not run on hardware
//
#include <hip/hip_runtime.h>

typedef _Float16 half_t;
typedef __attribute__((ext_vector_type(16))) _Float16 v16h;
typedef __attribute__((ext_vector_type(8)))  float    v8f;

#define DIM        32
#define NPTS       1024
#define TIME_SCALE 18.0f
#define LOG_M_OVER_HALF 0.575646273248511421f   /* ln(10000)/16 */

__global__ __launch_bounds__(256) void relds_wmma_kernel(
    const float* __restrict__ pos,   // [B, N, 3]
    const float* __restrict__ tim,   // [B, N]
    const float* __restrict__ W,     // [32, 32]  (rel = emb @ W^T + b)
    const float* __restrict__ bias,  // [32]
    float* __restrict__ rel_out,     // [B, N, N, 32]
    float* __restrict__ emb_out)     // [B, N, N, 32]
{
    const int lane  = (int)(threadIdx.x & 31u);
    const int wave  = (int)(threadIdx.x >> 5u);   // 0..7
    const int n     = (int)blockIdx.y;
    const int b     = (int)blockIdx.z;
    const int row16 = lane & 15;                  // M row within tile / N col of B
    const int hi    = lane >> 4;                  // 0: low half-lanes, 1: high

    // ---- per-(b,n) scalars (uniform across block) ----
    const float pnx = pos[((size_t)b * NPTS + n) * 3 + 0];
    const float pny = pos[((size_t)b * NPTS + n) * 3 + 1];
    const float pnz = pos[((size_t)b * NPTS + n) * 3 + 2];
    const float tn  = tim[(size_t)b * NPTS + n];

    // ---- per-lane frequency table: j = i + 8*hi, i = 0..7 ----
    float freq[8];
    const float jb = (float)(hi * 8);
#pragma unroll
    for (int i = 0; i < 8; ++i)
        freq[i] = __expf(-LOG_M_OVER_HALF * ((float)i + jb));

    // ---- B tiles: W^T in WMMA B-layout (f16), hoisted out of m-loop ----
    // B0 covers e = 0..15, B1 covers e = 16..31.
    // lane<16 : elem h holds B[K=h+(h&8)][N=lane]         -> W[e][h+(h&8)]
    // lane>=16: elem h holds B[K=8+h+(h&8)][N=lane-16]    -> W[e][8+h+(h&8)]
    v16h B0, B1;
    {
        const int kofs = hi ? 8 : 0;
#pragma unroll
        for (int h = 0; h < 16; ++h) {
            B0[h] = (half_t)W[(size_t)(row16)      * DIM + kofs + h + (h & 8)];
            B1[h] = (half_t)W[(size_t)(row16 + 16) * DIM + kofs + h + (h & 8)];
        }
    }

    // ---- C accumulators preloaded with bias (D-layout: N = lane%16 -> e) ----
    v8f C0, C1;
    {
        const float b0 = bias[row16];
        const float b1 = bias[row16 + 16];
#pragma unroll
        for (int r = 0; r < 8; ++r) { C0[r] = b0; C1[r] = b1; }
    }

    const size_t rowbase = ((size_t)b * NPTS + n) * NPTS;   // index of [b][n][0] in [B,N,N]

    for (int t = wave; t < NPTS / 16; t += 8) {
        const int m = t * 16 + row16;

        // ---- per-lane geometry for row m ----
        const float dx = pnx - pos[((size_t)b * NPTS + m) * 3 + 0];
        const float dy = pny - pos[((size_t)b * NPTS + m) * 3 + 1];
        const float dz = pnz - pos[((size_t)b * NPTS + m) * 3 + 2];
        const float dt = (tn - tim[(size_t)b * NPTS + m]) * TIME_SCALE;
        const float ds2 = dx * dx + dy * dy + dz * dz - dt * dt;
        float d = copysignf(__fsqrt_rn(fabsf(ds2)), ds2);
        d = fminf(4.0f, fmaxf(-4.0f, d));
        const float x = 1024.0f * d;

        // ---- 8 sin + 8 cos per lane; assembles A tile in WMMA A-layout ----
        float sv[8], cv[8];
#pragma unroll
        for (int i = 0; i < 8; ++i) {
            const float ang = x * freq[i];
            sv[i] = __sinf(ang);
            cv[i] = __cosf(ang);
        }
        v16h A;
#pragma unroll
        for (int i = 0; i < 8; ++i) {
            A[i]     = (half_t)sv[i];
            A[i + 8] = (half_t)cv[i];
        }

        // ---- D = A x B + bias : two 16x16x32 f16 WMMAs cover e = 0..31 ----
        v8f D0 = __builtin_amdgcn_wmma_f32_16x16x32_f16(
            false, A, false, B0, (short)0, C0, false, false);
        v8f D1 = __builtin_amdgcn_wmma_f32_16x16x32_f16(
            false, A, false, B1, (short)0, C1, false, false);

        // ---- emb store: this lane's K runs {kb..kb+7} and {kb+16..kb+23} ----
        {
            float* ep = emb_out + (rowbase + (size_t)m) * DIM + (hi ? 8 : 0);
            float4 s0 = make_float4(sv[0], sv[1], sv[2], sv[3]);
            float4 s1 = make_float4(sv[4], sv[5], sv[6], sv[7]);
            float4 c0 = make_float4(cv[0], cv[1], cv[2], cv[3]);
            float4 c1 = make_float4(cv[4], cv[5], cv[6], cv[7]);
            *(float4*)(ep +  0) = s0;
            *(float4*)(ep +  4) = s1;
            *(float4*)(ep + 16) = c0;
            *(float4*)(ep + 20) = c1;
        }

        // ---- rel_attn store: D-layout row r -> m = t*16 + 8*hi + r, e = row16 ----
        {
            const int mr = t * 16 + hi * 8;
#pragma unroll
            for (int r = 0; r < 8; ++r) {
                const size_t off = (rowbase + (size_t)(mr + r)) * DIM + row16;
                rel_out[off]      = D0[r];
                rel_out[off + 16] = D1[r];
            }
        }
    }
}

extern "C" void kernel_launch(void* const* d_in, const int* in_sizes, int n_in,
                              void* d_out, int out_size, void* d_ws, size_t ws_size,
                              hipStream_t stream) {
    const float* pos  = (const float*)d_in[0];   // [2,1024,3]
    const float* tim  = (const float*)d_in[1];   // [2,1024]
    const float* W    = (const float*)d_in[2];   // [32,32]
    const float* bias = (const float*)d_in[3];   // [32]

    float* rel = (float*)d_out;                                   // [2,1024,1024,32]
    float* emb = rel + (size_t)2 * NPTS * NPTS * DIM;             // [2,1024,1024,32]

    dim3 grid(1, NPTS, 2);   // 2048 blocks x 8 waves = 16384 wave-tasks
    dim3 block(256, 1, 1);
    relds_wmma_kernel<<<grid, block, 0, stream>>>(pos, tim, W, bias, rel, emb);
}
